// g2vec_26963804684786
// MI455X (gfx1250) — matrix operand order repaired and checked by reading, not compiled, not run on hardware
//
#include <hip/hip_runtime.h>
#include <hip/hip_bf16.h>

#define HD 128
#define ROWS_PB 64          // rows (M) per block in the WMMA GEMM
#define SLOPE 0.2f
#define NEG_BIG -3.0e38f

typedef _Float16 v16h __attribute__((ext_vector_type(16)));
typedef _Float16 v8h  __attribute__((ext_vector_type(8)));
typedef float    v8f  __attribute__((ext_vector_type(8)));

union AB16 { v16h v; v8h h[2]; _Float16 e[16]; };
union ACC8 { v8f v; float f[8]; };

__device__ __forceinline__ float leaky(float x) { return x > 0.f ? x : SLOPE * x; }

// Monotonic float atomic-max via signed/unsigned int atomics (finite values only).
__device__ __forceinline__ void atomicMaxF(float* addr, float v) {
  int iv = __float_as_int(v);
  if (iv >= 0) atomicMax((int*)addr, iv);
  else         atomicMin((unsigned int*)addr, (unsigned int)iv);
}

// ---------------------------------------------------------------------------
// Layer-1 GEMM: xp1[n,:] = f16( emb[h[n],:] ) @ f16(W1) via v_wmma_f32_16x16x32_f16
// Block = 256 threads (8 waves). Block covers 64 rows x 128 cols.
// Each wave owns one 16-col strip: loads its 4 B fragments once, then loops
// over 4 M-tiles (16 WMMAs per wave per block).
// ---------------------------------------------------------------------------
__global__ __launch_bounds__(256)
void gat_gemm1_wmma(const int* __restrict__ hidx,
                    const float* __restrict__ emb,
                    const float* __restrict__ W1,
                    float* __restrict__ xp1, int N)
{
  __shared__ __align__(16) _Float16 sWt[HD * HD];      // W1 transposed: sWt[n*HD+k] = W1[k,n]
  __shared__ __align__(16) _Float16 sA[ROWS_PB * HD];  // A tile (gathered via h), f16

  const int tid  = threadIdx.x;
  const int row0 = blockIdx.x * ROWS_PB;

  // Stage W1 (coalesced global read, transposed LDS write), f32 -> f16.
  for (int idx = tid; idx < HD * HD; idx += 256) {
    int k = idx >> 7, n = idx & (HD - 1);
    sWt[n * HD + k] = (_Float16)W1[idx];
  }
  // Stage A tile rows via embedding gather (clamped for the tail block).
  for (int idx = tid; idx < ROWS_PB * HD; idx += 256) {
    int r = idx >> 7, k = idx & (HD - 1);
    int node = row0 + r;
    int g = hidx[node < N ? node : (N - 1)];
    sA[idx] = (_Float16)emb[(size_t)g * HD + k];
  }
  __syncthreads();

  const int wave = tid >> 5, lane = tid & 31;
  const int m16 = lane & 15, hi = lane >> 4;

  // B fragments for this wave's 16-col strip, loaded once.
  // B 32x16 f16 layout: lane holds col n = wave*16+m16, K = ks+16*hi .. +15
  // (contiguous in the transposed W copy).
  AB16 b[4];
#pragma unroll
  for (int kb = 0; kb < 4; ++kb) {
    const _Float16* pb = &sWt[(wave * 16 + m16) * HD + kb * 32 + hi * 16];
    b[kb].h[0] = *(const v8h*)pb;
    b[kb].h[1] = *(const v8h*)(pb + 8);
  }

#pragma unroll
  for (int mt = 0; mt < ROWS_PB / 16; ++mt) {
    ACC8 acc;
#pragma unroll
    for (int i = 0; i < 8; ++i) acc.f[i] = 0.f;

#pragma unroll
    for (int kb = 0; kb < 4; ++kb) {
      // A 16x32 f16 layout: lane half 'hi' holds K = ks+16*hi..+15
      AB16 a;
      const _Float16* pa = &sA[(mt * 16 + m16) * HD + kb * 32 + hi * 16];
      a.h[0] = *(const v8h*)pa;
      a.h[1] = *(const v8h*)(pa + 8);
      acc.v = __builtin_amdgcn_wmma_f32_16x16x32_f16(false, a.v, false, b[kb].v,
                                                     (short)0, acc.v, false, false);
    }

    // D layout: VGPR r -> M = r + 8*hi, N = m16
    int base = row0 + mt * 16;
    if (base + 16 <= N) {
      float* dst = xp1 + (size_t)(base + hi * 8) * HD + wave * 16 + m16;
#pragma unroll
      for (int r = 0; r < 8; ++r) dst[(size_t)r * HD] = acc.f[r];
    } else {
#pragma unroll
      for (int r = 0; r < 8; ++r) {
        int node = base + hi * 8 + r;
        if (node < N) xp1[(size_t)node * HD + wave * 16 + m16] = acc.f[r];
      }
    }
  }
}

// ---------------------------------------------------------------------------
// Per-node attention scalars for layer 1: e_src, e_dst, self logit, m init.
// One wave32 per node.
// ---------------------------------------------------------------------------
__global__ __launch_bounds__(256)
void gat_node_attn1(const float* __restrict__ xp1,
                    const float* __restrict__ a_src, const float* __restrict__ a_dst,
                    float* __restrict__ es, float* __restrict__ ed,
                    float* __restrict__ selfl, float* __restrict__ m, int N)
{
  int wave = threadIdx.x >> 5, lane = threadIdx.x & 31;
  int node = blockIdx.x * 8 + wave;
  if (node >= N) return;
  const float* row = xp1 + (size_t)node * HD;
  float s = 0.f, d = 0.f;
#pragma unroll
  for (int i = 0; i < HD; i += 32) {
    float v = row[i + lane];
    s += v * a_src[i + lane];
    d += v * a_dst[i + lane];
  }
#pragma unroll
  for (int o = 16; o > 0; o >>= 1) {
    s += __shfl_xor(s, o, 32);
    d += __shfl_xor(d, o, 32);
  }
  if (lane == 0) {
    es[node] = s; ed[node] = d;
    float sl = leaky(s + d);
    selfl[node] = sl; m[node] = sl;
  }
}

// ---------------------------------------------------------------------------
// Per-edge masked logit + segment max (shared by both layers).
// ---------------------------------------------------------------------------
__global__ void gat_edge_logit_max(const int* __restrict__ ei, const int* __restrict__ nt,
                                   const float* __restrict__ es, const float* __restrict__ ed,
                                   float* __restrict__ elog, float* __restrict__ m, int E)
{
  int e = blockIdx.x * blockDim.x + threadIdx.x;
  if (e >= E) return;
  int s = ei[e], d = ei[E + e];
  float l = NEG_BIG;
  if ((nt[s] == 0) & (nt[d] == 0)) {
    l = leaky(es[s] + ed[d]);
    atomicMaxF(&m[d], l);
  }
  elog[e] = l;
}

// Per-node: self weight + denom init (shared by both layers).
__global__ void gat_node_sw(const float* __restrict__ selfl, const float* __restrict__ m,
                            float* __restrict__ sw, float* __restrict__ den, int N)
{
  int i = blockIdx.x * blockDim.x + threadIdx.x;
  if (i >= N) return;
  float s = __expf(selfl[i] - m[i]);
  sw[i] = s; den[i] = s;
}

// Per-edge: ew = exp(logit - m[dst]); denom += ew; cache ew in place of logit.
__global__ void gat_edge_expsum(const int* __restrict__ ei, const float* __restrict__ m,
                                float* __restrict__ elog, float* __restrict__ den, int E)
{
  int e = blockIdx.x * blockDim.x + threadIdx.x;
  if (e >= E) return;
  float l = elog[e];
  float w = 0.f;
  if (l > -1.0e38f) {
    int d = ei[E + e];
    w = __expf(l - m[d]);
    atomicAdd(&den[d], w);
  }
  elog[e] = w;
}

// Layer-1 aggregation: 32 lanes per edge, float4 per lane; atomics land in L2.
__global__ void gat_edge_aggregate(const int* __restrict__ ei, const float* __restrict__ ew,
                                   const float* __restrict__ den, const float* __restrict__ xp,
                                   float* __restrict__ acc, int E)
{
  int gt = blockIdx.x * blockDim.x + threadIdx.x;
  int e = gt >> 5, lane = gt & 31;
  if (e >= E) return;
  float w = ew[e];
  if (w == 0.f) return;
  int s = ei[e], d = ei[E + e];
  float alpha = w / den[d];
  const float4 v = *(const float4*)(xp + (size_t)s * HD + lane * 4);
  float* p = acc + (size_t)d * HD + lane * 4;
  atomicAdd(p + 0, alpha * v.x);
  atomicAdd(p + 1, alpha * v.y);
  atomicAdd(p + 2, alpha * v.z);
  atomicAdd(p + 3, alpha * v.w);
}

// Layer-1 finalize: add self-loop term + bias, ReLU (in place on accumulator).
__global__ void gat_finalize1(float* __restrict__ acc, const float* __restrict__ xp1,
                              const float* __restrict__ sw, const float* __restrict__ den,
                              const float* __restrict__ b, int total)
{
  int i = blockIdx.x * blockDim.x + threadIdx.x;
  if (i >= total) return;
  int node = i >> 7, f = i & (HD - 1);
  float v = acc[i] + (sw[node] / den[node]) * xp1[i] + b[f];
  acc[i] = v > 0.f ? v : 0.f;
}

// Layer-2 projection (OUT=1): xp2 = out1 . W2, plus scalar attention setup.
__global__ __launch_bounds__(256)
void gat_node_xp2(const float* __restrict__ out1, const float* __restrict__ W2,
                  const float* __restrict__ as2, const float* __restrict__ ad2,
                  float* __restrict__ xp2, float* __restrict__ es, float* __restrict__ ed,
                  float* __restrict__ selfl, float* __restrict__ m, int N)
{
  int wave = threadIdx.x >> 5, lane = threadIdx.x & 31;
  int node = blockIdx.x * 8 + wave;
  if (node >= N) return;
  const float* row = out1 + (size_t)node * HD;
  float acc = 0.f;
#pragma unroll
  for (int i = 0; i < HD; i += 32) acc += row[i + lane] * W2[i + lane];
#pragma unroll
  for (int o = 16; o > 0; o >>= 1) acc += __shfl_xor(acc, o, 32);
  if (lane == 0) {
    xp2[node] = acc;
    float s = acc * as2[0], d = acc * ad2[0];
    es[node] = s; ed[node] = d;
    float sl = leaky(s + d);
    selfl[node] = sl; m[node] = sl;
  }
}

// Layer-2 aggregation: single feature, one atomic per edge.
__global__ void gat_edge_aggregate2(const int* __restrict__ ei, const float* __restrict__ ew,
                                    const float* __restrict__ den, const float* __restrict__ xp2,
                                    float* __restrict__ out, int E)
{
  int e = blockIdx.x * blockDim.x + threadIdx.x;
  if (e >= E) return;
  float w = ew[e];
  if (w == 0.f) return;
  int s = ei[e], d = ei[E + e];
  atomicAdd(&out[d], (w / den[d]) * xp2[s]);
}

__global__ void gat_finalize2(float* __restrict__ out, const float* __restrict__ xp2,
                              const float* __restrict__ sw, const float* __restrict__ den,
                              const float* __restrict__ b2, int N)
{
  int i = blockIdx.x * blockDim.x + threadIdx.x;
  if (i >= N) return;
  out[i] += (sw[i] / den[i]) * xp2[i] + b2[0];
}

// ---------------------------------------------------------------------------
extern "C" void kernel_launch(void* const* d_in, const int* in_sizes, int n_in,
                              void* d_out, int out_size, void* d_ws, size_t ws_size,
                              hipStream_t stream) {
  (void)n_in; (void)out_size; (void)ws_size;
  const int*   hidx = (const int*)d_in[0];
  const int*   ei   = (const int*)d_in[1];
  const int*   nt   = (const int*)d_in[2];
  const float* emb  = (const float*)d_in[3];
  const float* W1   = (const float*)d_in[4];
  const float* as1  = (const float*)d_in[5];
  const float* ad1  = (const float*)d_in[6];
  const float* b1   = (const float*)d_in[7];
  const float* W2   = (const float*)d_in[8];
  const float* as2  = (const float*)d_in[9];
  const float* ad2  = (const float*)d_in[10];
  const float* b2   = (const float*)d_in[11];
  float* out = (float*)d_out;

  const int N = in_sizes[0];
  const int E = in_sizes[1] / 2;

  // Scratch carve-up (floats). Node-scalar buffers reused between layers.
  float* p = (float*)d_ws;
  float* xp1   = p; p += (size_t)N * HD;
  float* acc1  = p; p += (size_t)N * HD;
  float* es    = p; p += N;
  float* ed    = p; p += N;
  float* selfl = p; p += N;
  float* m     = p; p += N;
  float* sw    = p; p += N;
  float* den   = p; p += N;
  float* xp2   = p; p += N;
  float* elog  = p; p += E;   // logits, then exp-weights, reused per layer

  const int TB = 256;
  const int eB = (E + TB - 1) / TB;
  const int nB = (N + TB - 1) / TB;

  // ---- Layer 1 ----
  gat_gemm1_wmma<<<(N + ROWS_PB - 1) / ROWS_PB, TB, 0, stream>>>(hidx, emb, W1, xp1, N);
  gat_node_attn1<<<(N + 7) / 8, TB, 0, stream>>>(xp1, as1, ad1, es, ed, selfl, m, N);
  gat_edge_logit_max<<<eB, TB, 0, stream>>>(ei, nt, es, ed, elog, m, E);
  gat_node_sw<<<nB, TB, 0, stream>>>(selfl, m, sw, den, N);
  gat_edge_expsum<<<eB, TB, 0, stream>>>(ei, m, elog, den, E);
  hipMemsetAsync(acc1, 0, (size_t)N * HD * sizeof(float), stream);
  {
    long long tot = (long long)E * 32;
    int blocks = (int)((tot + TB - 1) / TB);
    gat_edge_aggregate<<<blocks, TB, 0, stream>>>(ei, elog, den, xp1, acc1, E);
  }
  {
    int total = N * HD;
    gat_finalize1<<<(total + TB - 1) / TB, TB, 0, stream>>>(acc1, xp1, sw, den, b1, total);
  }

  // ---- Layer 2 (OUT = 1) ----
  gat_node_xp2<<<(N + 7) / 8, TB, 0, stream>>>(acc1, W2, as2, ad2, xp2, es, ed, selfl, m, N);
  gat_edge_logit_max<<<eB, TB, 0, stream>>>(ei, nt, es, ed, elog, m, E);
  gat_node_sw<<<nB, TB, 0, stream>>>(selfl, m, sw, den, N);
  gat_edge_expsum<<<eB, TB, 0, stream>>>(ei, m, elog, den, E);
  hipMemsetAsync(out, 0, (size_t)N * sizeof(float), stream);
  gat_edge_aggregate2<<<eB, TB, 0, stream>>>(ei, elog, den, xp2, out, E);
  gat_finalize2<<<nB, TB, 0, stream>>>(out, xp2, sw, den, b2, N);
}
